// LSTM_Net_83958020702886
// MI455X (gfx1250) — compile-verified, matrix operand order checked by reading, not run on hardware
//
#include <hip/hip_runtime.h>
#include <hip/hip_bf16.h>

typedef __attribute__((ext_vector_type(16))) __bf16 v16bf;
typedef __attribute__((ext_vector_type(8)))  __bf16 v8bf;
typedef __attribute__((ext_vector_type(8)))  float  v8f;

#define BATCH 16384
#define SEQT  50
#define HID   50
#define LIN   8
#define WAVES 8
#define KP    64      // padded K (hidden) dimension
#define NT    16      // N tiles: 4 gates x 4 j-tiles of 16 (each gate padded 50->64)

// --- LDS partition (bytes), total 134656 ---
#define OFF_FR    0            // 3 weights x 16384 bf16 fragments = 98304 B
#define OFF_HLDS  98304        // 8 waves x 2 layers x 16 x 64 bf16 = 32768 B
#define OFF_BP0   131072       // 256 f32
#define OFF_BP1   132096       // 256 f32
#define OFF_W0P   133120       // 256 f32
#define OFF_XL    134144       // 8 x 16 f32 = 512 B
#define SMEM_BYTES 134656

__device__ __forceinline__ float fsigmoid(float x) { return 1.0f / (1.0f + __expf(-x)); }
__device__ __forceinline__ float ftanh(float x) { float e = __expf(2.0f * x); return (e - 1.0f) / (e + 1.0f); }

__device__ __forceinline__ v8f wmma_bf16(v16bf a, v16bf b, v8f c) {
  return __builtin_amdgcn_wmma_f32_16x16x32_bf16(false, a, false, b, (short)0, c, false, false);
}

// A-matrix fragment (16x32 bf16) per ISA 7.12.2: lane m=l%16, group g=l/16,
// element 2v+half <-> K = 32*kt + 16*(v/4) + 8*g + 2*(v%4) + half.
// With h stored row-major [m][64], each fragment is two contiguous 16B reads.
__device__ __forceinline__ void load_afrag(const __bf16* hL, int lane, v16bf a[2]) {
  const int m = lane & 15, g = lane >> 4;
  const __bf16* row = hL + m * KP;
#pragma unroll
  for (int kt = 0; kt < 2; ++kt) {
    v8bf lo = *reinterpret_cast<const v8bf*>(row + 32 * kt + 8 * g);
    v8bf hi = *reinterpret_cast<const v8bf*>(row + 32 * kt + 16 + 8 * g);
    a[kt] = __builtin_shufflevector(lo, hi, 0,1,2,3,4,5,6,7,8,9,10,11,12,13,14,15);
  }
}

// B fragments pre-swizzled in LDS: one contiguous 32B read per lane.
__device__ __forceinline__ v16bf load_bfrag(const __bf16* fr, int nt, int kt, int lane) {
  return *reinterpret_cast<const v16bf*>(fr + (((nt << 1) + kt) * 32 + lane) * 16);
}

__global__ void __launch_bounds__(256, 1) lstm2_wmma_kernel(
    const float* __restrict__ x,
    const float* __restrict__ h0in, const float* __restrict__ c0in,
    const float* __restrict__ wih0, const float* __restrict__ whh0,
    const float* __restrict__ bih0, const float* __restrict__ bhh0,
    const float* __restrict__ wih1, const float* __restrict__ whh1,
    const float* __restrict__ bih1, const float* __restrict__ bhh1,
    const float* __restrict__ wl1,  const float* __restrict__ bl1,
    const float* __restrict__ wl2,  const float* __restrict__ bl2,
    float* __restrict__ out)
{
  extern __shared__ char smem[];
  __bf16* frbase = reinterpret_cast<__bf16*>(smem + OFF_FR);
  __bf16* fr_hh0 = frbase;
  __bf16* fr_ih1 = frbase + 16384;
  __bf16* fr_hh1 = frbase + 32768;
  __bf16* hlds   = reinterpret_cast<__bf16*>(smem + OFF_HLDS);
  float*  bp0    = reinterpret_cast<float*>(smem + OFF_BP0);
  float*  bp1    = reinterpret_cast<float*>(smem + OFF_BP1);
  float*  w0p    = reinterpret_cast<float*>(smem + OFF_W0P);
  float*  xlds   = reinterpret_cast<float*>(smem + OFF_XL);

  const int tid  = threadIdx.x;
  const int wave = tid >> 5;
  const int lane = tid & 31;
  const int g    = lane >> 4;
  const int n16  = lane & 15;
  const int rowbase = (blockIdx.x * WAVES + wave) * 16;

  // ---- One-time: swizzle weights into WMMA B-fragment layout (bf16) ----
  // Gate-column permutation: padded col = nt*16+n16; gate q = col/64, j = col%64.
  for (int idx = tid; idx < 3 * 16384; idx += blockDim.x) {
    int w  = idx >> 14;
    int r  = idx & 16383;
    int hh   = r & 15;
    int ln   = (r >> 4) & 31;
    int ktnt = r >> 9;          // (nt*2 + kt)
    int kt = ktnt & 1, nt = ktnt >> 1;
    int gg = ln >> 4;
    int col = nt * 16 + (ln & 15);
    int q = col >> 6, j = col & 63;
    int v = hh >> 1, half = hh & 1;
    int k = 32 * kt + 16 * (v >> 2) + 8 * gg + 2 * (v & 3) + half;
    const float* W = (w == 0) ? whh0 : ((w == 1) ? wih1 : whh1);
    float val = (j < HID && k < HID) ? W[(q * HID + j) * HID + k] : 0.0f;
    frbase[idx] = (__bf16)val;
  }
  // Permuted fused biases + layer-0 input weights (IN_DIM=1 column vector)
  for (int col = tid; col < 256; col += blockDim.x) {
    int q = col >> 6, j = col & 63;
    bp0[col] = (j < HID) ? (bih0[q * HID + j] + bhh0[q * HID + j]) : 0.0f;
    bp1[col] = (j < HID) ? (bih1[q * HID + j] + bhh1[q * HID + j]) : 0.0f;
    w0p[col] = (j < HID) ? wih0[q * HID + j] : 0.0f;
  }

  // ---- Per-wave state init (h in LDS bf16, c in f32 registers) ----
  __bf16* h0L = hlds + (wave * 2 + 0) * 16 * KP;
  __bf16* h1L = hlds + (wave * 2 + 1) * 16 * KP;
  v8f c0s[4], c1s[4];
#pragma unroll
  for (int jt = 0; jt < 4; ++jt) {
    int j = jt * 16 + n16;
#pragma unroll
    for (int v = 0; v < 8; ++v) {
      int m = v + 8 * g, b = rowbase + m;
      float h0v = 0.f, h1v = 0.f, cc0 = 0.f, cc1 = 0.f;
      if (j < HID) {
        h0v = h0in[0 * BATCH * HID + b * HID + j];
        h1v = h0in[1 * BATCH * HID + b * HID + j];
        cc0 = c0in[0 * BATCH * HID + b * HID + j];
        cc1 = c0in[1 * BATCH * HID + b * HID + j];
      }
      c0s[jt][v] = cc0;  c1s[jt][v] = cc1;
      h0L[m * KP + j] = (__bf16)h0v;
      h1L[m * KP + j] = (__bf16)h1v;
    }
  }
  __syncthreads();

  float* outPred = out;
  float* outH = out + BATCH;
  float* outC = out + BATCH + 2 * BATCH * HID;

  // ---- Recurrence over T=50 timesteps ----
  for (int t = 0; t < SEQT; ++t) {
    const bool last = (t == SEQT - 1);

    // stage x_t (scaled) for this wave's 16 rows
    if (lane < 16) xlds[wave * 16 + lane] = x[(rowbase + lane) * SEQT + t] * (1.0f / 1.5f);
    float xa[8];
#pragma unroll
    for (int v = 0; v < 8; ++v) xa[v] = xlds[wave * 16 + v + 8 * g];

    // ===== Layer 0: gates = x*w_ih0 + h0 @ Whh0^T + b =====
    {
      v16bf a1[2];
      load_afrag(h0L, lane, a1);
#pragma unroll
      for (int jt = 0; jt < 4; ++jt) {
        v8f acc[4];
#pragma unroll
        for (int q = 0; q < 4; ++q) {
          int nt = q * 4 + jt;
          float bb = bp0[nt * 16 + n16];
          float w0 = w0p[nt * 16 + n16];
#pragma unroll
          for (int v = 0; v < 8; ++v) acc[q][v] = bb + xa[v] * w0;
#pragma unroll
          for (int kt = 0; kt < 2; ++kt)
            acc[q] = wmma_bf16(a1[kt], load_bfrag(fr_hh0, nt, kt, lane), acc[q]);
        }
#pragma unroll
        for (int v = 0; v < 8; ++v) {
          float ii = fsigmoid(acc[0][v]);
          float ff = fsigmoid(acc[1][v]);
          float gg = ftanh(acc[2][v]);
          float oo = fsigmoid(acc[3][v]);
          float cn = ff * c0s[jt][v] + ii * gg;
          c0s[jt][v] = cn;
          float hn = oo * ftanh(cn);
          int m = v + 8 * g, j = jt * 16 + n16;
          h0L[m * KP + j] = (__bf16)hn;
          if (last && j < HID) {
            int b = rowbase + m;
            outH[0 * BATCH * HID + b * HID + j] = hn;
            outC[0 * BATCH * HID + b * HID + j] = cn;
          }
        }
      }
    }

    // ===== Layer 1: gates = h0(t) @ Wih1^T + h1 @ Whh1^T + b =====
    {
      v16bf a0[2], a1[2];
      load_afrag(h0L, lane, a0);   // fresh layer-0 output (same timestep)
      load_afrag(h1L, lane, a1);   // previous h1
#pragma unroll
      for (int jt = 0; jt < 4; ++jt) {
        v8f acc[4];
#pragma unroll
        for (int q = 0; q < 4; ++q) {
          int nt = q * 4 + jt;
          float bb = bp1[nt * 16 + n16];
#pragma unroll
          for (int v = 0; v < 8; ++v) acc[q][v] = bb;
#pragma unroll
          for (int kt = 0; kt < 2; ++kt) {
            acc[q] = wmma_bf16(a0[kt], load_bfrag(fr_ih1, nt, kt, lane), acc[q]);
            acc[q] = wmma_bf16(a1[kt], load_bfrag(fr_hh1, nt, kt, lane), acc[q]);
          }
        }
#pragma unroll
        for (int v = 0; v < 8; ++v) {
          float ii = fsigmoid(acc[0][v]);
          float ff = fsigmoid(acc[1][v]);
          float gg = ftanh(acc[2][v]);
          float oo = fsigmoid(acc[3][v]);
          float cn = ff * c1s[jt][v] + ii * gg;
          c1s[jt][v] = cn;
          float hn = oo * ftanh(cn);
          int m = v + 8 * g, j = jt * 16 + n16;
          h1L[m * KP + j] = (__bf16)hn;
          if (last && j < HID) {
            int b = rowbase + m;
            outH[1 * BATCH * HID + b * HID + j] = hn;
            outC[1 * BATCH * HID + b * HID + j] = cn;
          }
        }
      }
    }
  }

  // ---- MLP head on last h1: softsign(h1 @ Wl1^T + b1) @ Wl2^T + b2, x70 ----
  if (lane < 16) {
    int m = lane, b = rowbase + m;
    float pr = bl2[0];
#pragma unroll
    for (int p = 0; p < LIN; ++p) {
      float s = bl1[p];
      for (int k = 0; k < HID; ++k)
        s += (float)h1L[m * KP + k] * wl1[p * HID + k];
      float ss = s / (1.0f + fabsf(s));
      pr += ss * wl2[p];
    }
    outPred[b] = pr * 70.0f;
  }
}

extern "C" void kernel_launch(void* const* d_in, const int* in_sizes, int n_in,
                              void* d_out, int out_size, void* d_ws, size_t ws_size,
                              hipStream_t stream) {
  (void)in_sizes; (void)n_in; (void)out_size; (void)d_ws; (void)ws_size;
  const float* x    = (const float*)d_in[0];
  const float* h0   = (const float*)d_in[1];
  const float* c0   = (const float*)d_in[2];
  const float* wih0 = (const float*)d_in[3];
  const float* whh0 = (const float*)d_in[4];
  const float* bih0 = (const float*)d_in[5];
  const float* bhh0 = (const float*)d_in[6];
  const float* wih1 = (const float*)d_in[7];
  const float* whh1 = (const float*)d_in[8];
  const float* bih1 = (const float*)d_in[9];
  const float* bhh1 = (const float*)d_in[10];
  const float* wl1  = (const float*)d_in[11];
  const float* bl1  = (const float*)d_in[12];
  const float* wl2  = (const float*)d_in[13];
  const float* bl2  = (const float*)d_in[14];
  float* out = (float*)d_out;

  dim3 grid(BATCH / (WAVES * 16));   // 128 blocks
  dim3 block(WAVES * 32);            // 256 threads = 8 waves
  lstm2_wmma_kernel<<<grid, block, SMEM_BYTES, stream>>>(
      x, h0, c0, wih0, whh0, bih0, bhh0, wih1, whh1, bih1, bhh1,
      wl1, bl1, wl2, bl2, out);
}
